// ContrastiveLoss_43198781063230
// MI455X (gfx1250) — matrix operand order repaired and checked by reading, not compiled, not run on hardware
//
#include <hip/hip_runtime.h>
#include <hip/hip_bf16.h>
#include <math.h>

typedef __attribute__((ext_vector_type(16))) __bf16 v16bf;
typedef __attribute__((ext_vector_type(8)))  float  v8f;

#define D_DIM  1024
#define NROWS  4096
#define TILE_M 128
#define TILE_N 128
#define TILE_K 32
#define LDSTR  40   // 32 + 8 halves pad: 80B row stride, 16B aligned, conflict-free

__device__ static inline unsigned short f32_to_bf16(float f) {
    union { float f; unsigned u; } c; c.f = f;
    unsigned u = c.u;
    unsigned r = u + 0x7FFFu + ((u >> 16) & 1u);   // round-to-nearest-even
    return (unsigned short)(r >> 16);
}

// Wave-relative LDS byte offset of a shared-memory pointer (VGPR operand for
// the async-to-LDS instructions, per VFLAT VDST semantics).
__device__ static inline unsigned lds_addr_of(const void* p) {
    return (unsigned)(unsigned long long)
        (__attribute__((address_space(3))) const void*)p;
}

// One block per row: L2-normalize in fp32, emit bf16 row into workspace.
__global__ __launch_bounds__(256)
void normalize_rows_kernel(const float* __restrict__ in,
                           unsigned short* __restrict__ out) {
    const int row = blockIdx.x;
    const float* r = in + (size_t)row * D_DIM;
    float s = 0.f;
    for (int i = threadIdx.x; i < D_DIM; i += 256) { float v = r[i]; s += v * v; }
    for (int off = 16; off > 0; off >>= 1) s += __shfl_down(s, off, 32);
    __shared__ float red[8];
    const int wave = threadIdx.x >> 5, lane = threadIdx.x & 31;
    if (lane == 0) red[wave] = s;
    __syncthreads();
    if (threadIdx.x == 0) {
        float t = 0.f;
        for (int w = 0; w < 8; ++w) t += red[w];
        red[0] = 1.0f / fmaxf(sqrtf(t), 1e-8f);
    }
    __syncthreads();
    const float inv = red[0];
    unsigned short* o = out + (size_t)row * D_DIM;
    for (int i = threadIdx.x; i < D_DIM; i += 256) o[i] = f32_to_bf16(r[i] * inv);
}

// Issue async global->LDS copies for one 128x32 bf16 tile of A and of B.
// 256 threads x 2 passes x 16B = 8KB per tile; no VGPR staging, tracked by
// ASYNCcnt (CDNA5 async-to-LDS path).
__device__ static inline void async_tile_loads(
    const unsigned short* __restrict__ A, const unsigned short* __restrict__ B,
    unsigned short* As, unsigned short* Bs,
    int i0, int j0, int k0, int tid)
{
#pragma unroll
    for (int p = 0; p < 2; ++p) {
        int idx = tid + p * 256;            // 0..511
        int row = idx >> 2;                 // 0..127
        int col = (idx & 3) * 8;            // 0,8,16,24 halves (16B chunks)
        const unsigned short* ga = A + (size_t)(i0 + row) * D_DIM + k0 + col;
        unsigned la = lds_addr_of(As + row * LDSTR + col);
        asm volatile("global_load_async_to_lds_b128 %0, %1, off"
                     :: "v"(la), "v"(ga) : "memory");
        const unsigned short* gb = B + (size_t)(j0 + row) * D_DIM + k0 + col;
        unsigned lb = lds_addr_of(Bs + row * LDSTR + col);
        asm volatile("global_load_async_to_lds_b128 %0, %1, off"
                     :: "v"(lb), "v"(gb) : "memory");
    }
}

// Fused GEMM (sim = A * B^T over normalized bf16 rows) + exp / exp(s^2)
// global reduction + export of sim row 0. Double-buffered LDS fed by the
// async Tensor/LDS DMA path; never materializes the 4096^2 sim matrix.
__global__ __launch_bounds__(256)
void wmma_sim_reduce_kernel(const unsigned short* __restrict__ A,
                            const unsigned short* __restrict__ B,
                            float* __restrict__ out_row,
                            double* __restrict__ accum,
                            int mode /* 0: exp(s), 1: exp(s*s) */) {
    __shared__ unsigned short As[2][TILE_M * LDSTR];
    __shared__ unsigned short Bs[2][TILE_N * LDSTR];

    const int tid  = threadIdx.x;
    const int lane = tid & 31;
    const int wave = tid >> 5;
    const int wm   = wave & 1;     // 0..1 : 64-row strip of the 128x128 tile
    const int wn   = wave >> 1;    // 0..3 : 32-col strip
    const int half = lane >> 4;    // 0/1 (lane group per ISA WMMA layout)
    const int l16  = lane & 15;

    const int i0 = blockIdx.y * TILE_M;   // anchor rows
    const int j0 = blockIdx.x * TILE_N;   // positive/negative rows

    const v8f zero = {0.f, 0.f, 0.f, 0.f, 0.f, 0.f, 0.f, 0.f};
    v8f c[4][2];
#pragma unroll
    for (int a = 0; a < 4; ++a)
#pragma unroll
        for (int b = 0; b < 2; ++b) c[a][b] = zero;

    union Frag { v16bf v; uint4 q[2]; };

    // Prologue: fill buffer 0, publish to all waves.
    async_tile_loads(A, B, As[0], Bs[0], i0, j0, 0, tid);
    asm volatile("s_wait_asynccnt 0" ::: "memory");
    __syncthreads();

    for (int k0 = 0; k0 < D_DIM; k0 += TILE_K) {
        const int buf  = (k0 / TILE_K) & 1;
        const bool more = (k0 + TILE_K) < D_DIM;

        // Prefetch next K-tile into the other buffer while computing this one.
        // Safe: the barrier ending the previous iteration guarantees every
        // wave finished its ds_load reads of buf^1.
        if (more)
            async_tile_loads(A, B, As[buf ^ 1], Bs[buf ^ 1],
                             i0, j0, k0 + TILE_K, tid);

        // A fragment (16x32, ISA 16-bit A layout): lane<16 -> M=l16, K 0-7 &
        // 16-23; lane>=16 -> K 8-15 & 24-31.
        Frag af[4];
#pragma unroll
        for (int a = 0; a < 4; ++a) {
            const unsigned short* base =
                &As[buf][(wm * 64 + a * 16 + l16) * LDSTR] + half * 8;
            af[a].q[0] = *(const uint4*)(base);        // K = half*8 .. +7
            af[a].q[1] = *(const uint4*)(base + 16);   // K = 16+half*8 .. +7
        }
        // B fragment (32x16): column N = row of b (row-major b == column-major
        // B); lane holds K = half*8 .. +7 and 16+half*8 .. +7 (same as A).
        Frag bf[2];
#pragma unroll
        for (int b = 0; b < 2; ++b) {
            const unsigned short* base =
                &Bs[buf][(wn * 32 + b * 16 + l16) * LDSTR] + half * 8;
            bf[b].q[0] = *(const uint4*)(base);
            bf[b].q[1] = *(const uint4*)(base + 16);
        }
#pragma unroll
        for (int a = 0; a < 4; ++a)
#pragma unroll
            for (int b = 0; b < 2; ++b)
                c[a][b] = __builtin_amdgcn_wmma_f32_16x16x32_bf16(
                    false, af[a].v, false, bf[b].v, (short)0, c[a][b],
                    false, false);

        if (more) asm volatile("s_wait_asynccnt 0" ::: "memory");
        __syncthreads();
    }

    // Epilogue: C layout = VGPR v holds M = v (lanes 0-15) / v+8 (lanes 16-31),
    // N = lane%16. Fuse exp-reduction; export global row 0 of sim.
    float acc = 0.f;
#pragma unroll
    for (int a = 0; a < 4; ++a) {
#pragma unroll
        for (int b = 0; b < 2; ++b) {
#pragma unroll
            for (int v = 0; v < 8; ++v) {
                float s = c[a][b][v];
                acc += (mode == 0) ? expf(s) : expf(s * s);
                int gi = i0 + wm * 64 + a * 16 + v + half * 8;
                if (gi == 0) {
                    int gj = j0 + wn * 32 + b * 16 + l16;
                    out_row[gj] = s;
                }
            }
        }
    }
    for (int off = 16; off > 0; off >>= 1) acc += __shfl_down(acc, off, 32);
    __shared__ float red[8];
    if (lane == 0) red[wave] = acc;
    __syncthreads();
    if (tid == 0) {
        float t = 0.f;
        for (int w = 0; w < 8; ++w) t += red[w];
        atomicAdd(accum, (double)t);
    }
}

__global__ void zero_accum_kernel(double* acc) {
    if (threadIdx.x < 2) acc[threadIdx.x] = 0.0;
}

__global__ void finalize_kernel(const double* __restrict__ acc,
                                float* __restrict__ out) {
    const double inv = 1.0 / (4096.0 * 4096.0);
    double ep = acc[0] * inv;
    double en = acc[1] * inv;
    out[0] = (float)(-log(ep / (ep + en)));
}

extern "C" void kernel_launch(void* const* d_in, const int* in_sizes, int n_in,
                              void* d_out, int out_size, void* d_ws, size_t ws_size,
                              hipStream_t stream) {
    (void)in_sizes; (void)n_in; (void)out_size; (void)ws_size;
    const float* anchor   = (const float*)d_in[0];
    const float* positive = (const float*)d_in[1];
    const float* negative = (const float*)d_in[2];
    float* out = (float*)d_out;   // [loss, sim_pos[0] (4096), sim_neg[0] (4096)]

    unsigned short* An = (unsigned short*)d_ws;                 // 8 MB
    unsigned short* Pn = An + (size_t)NROWS * D_DIM;            // 8 MB
    unsigned short* Nn = Pn + (size_t)NROWS * D_DIM;            // 8 MB
    double* acc = (double*)(Nn + (size_t)NROWS * D_DIM);        // 2 doubles

    normalize_rows_kernel<<<NROWS, 256, 0, stream>>>(anchor,   An);
    normalize_rows_kernel<<<NROWS, 256, 0, stream>>>(positive, Pn);
    normalize_rows_kernel<<<NROWS, 256, 0, stream>>>(negative, Nn);
    zero_accum_kernel<<<1, 32, 0, stream>>>(acc);

    dim3 grid(NROWS / TILE_N, NROWS / TILE_M);
    wmma_sim_reduce_kernel<<<grid, 256, 0, stream>>>(An, Pn, out + 1,        acc + 0, 0);
    wmma_sim_reduce_kernel<<<grid, 256, 0, stream>>>(An, Nn, out + 1 + 4096, acc + 1, 1);

    finalize_kernel<<<1, 1, 0, stream>>>(acc, out);
}
